// Avey_72327249265141
// MI455X (gfx1250) — compile-verified, matrix operand order checked by reading, not run on hardware
//
#include <hip/hip_runtime.h>
#include <hip/hip_bf16.h>
#include <math.h>

typedef __bf16 bf16;
typedef __attribute__((ext_vector_type(16))) __bf16 v16bf;
typedef __attribute__((ext_vector_type(8)))  __bf16 v8bf;
typedef __attribute__((ext_vector_type(8)))  float  v8f;

#define E_    768
#define ED_   3072
#define S0_   1536
#define PROJ_ 2304
#define TL_   1024
#define NIMG_ 16          // B*N images
#define ROWS_ (NIMG_*TL_) // 16384
#define BT_   4096        // B*T
#define V_    32000
#define L_    4
#define C_    256
#define N_    8
#define B_    2
#define NPAIR_ 56         // B * 28 chunk pairs

// ---------------------------------------------------------------------------
// WMMA fragment helpers (bf16 16x16x32, f32 accumulate). Wave32.
// A-fragment layout (ISA 7.12.2, 16-bit A 16x32):
//   lanes 0-15 : row M=lane,     v0..v3 = K[0..7],  v4..v7 = K[16..23]
//   lanes 16-31: row M=lane-16,  v0..v3 = K[8..15], v4..v7 = K[24..31]
// B fragment for C = X @ W^T with row-major W(N,K): identical pattern on W rows.
// lanebase = &M[row + lane%16][ (lane/16)*8 ]; k0 folds into load immediates.
// ---------------------------------------------------------------------------
__device__ __forceinline__ v16bf frag_ld(const bf16* lanebase, int k0) {
  v8bf lo = *(const v8bf*)(lanebase + k0);        // K block 0 (16B)
  v8bf hi = *(const v8bf*)(lanebase + k0 + 16);   // K block +16
  v16bf out;
#pragma unroll
  for (int i = 0; i < 8; ++i) { out[i] = lo[i]; out[8 + i] = hi[i]; }
  return out;
}

#define WMMA_BF16(a, b, c) __builtin_amdgcn_wmma_f32_16x16x32_bf16( \
    false, (a), false, (b), (short)0, (c), false, false)

// 32(M) x 64(N) per-wave tile, K stepped by 32. Six fragments (2A + 4B) are
// issued as one clause, then 8 WMMAs; ~172 live VGPRs leaves the scheduler
// room to overlap next-iteration loads without spilling.
__device__ __forceinline__ void gemm_core_2x4(const bf16* A, int lda,
                                              const bf16* W, int ldw,
                                              int row0, int col0, int K,
                                              v8f acc[2][4]) {
  int lane = threadIdx.x & 31;
  int l15 = lane & 15, s8 = (lane >> 4) * 8;
  const bf16* a0 = A + (size_t)(row0 + l15) * lda + s8;
  const bf16* a1 = a0 + (size_t)16 * lda;
  const bf16* b0 = W + (size_t)(col0 + l15) * ldw + s8;
  const bf16* b1 = b0 + (size_t)16 * ldw;
  const bf16* b2 = b1 + (size_t)16 * ldw;
  const bf16* b3 = b2 + (size_t)16 * ldw;
  for (int k0 = 0; k0 < K; k0 += 32) {
    v16bf af0 = frag_ld(a0, k0);
    v16bf af1 = frag_ld(a1, k0);
    v16bf bf0 = frag_ld(b0, k0);
    v16bf bf1 = frag_ld(b1, k0);
    v16bf bf2 = frag_ld(b2, k0);
    v16bf bf3 = frag_ld(b3, k0);
    if (k0 + 32 < K)  // pull next A slab toward L2 (global_prefetch_b8)
      __builtin_prefetch((const void*)(a0 + k0 + 32), 0, 1);
    acc[0][0] = WMMA_BF16(af0, bf0, acc[0][0]);
    acc[1][0] = WMMA_BF16(af1, bf0, acc[1][0]);
    acc[0][1] = WMMA_BF16(af0, bf1, acc[0][1]);
    acc[1][1] = WMMA_BF16(af1, bf1, acc[1][1]);
    acc[0][2] = WMMA_BF16(af0, bf2, acc[0][2]);
    acc[1][2] = WMMA_BF16(af1, bf2, acc[1][2]);
    acc[0][3] = WMMA_BF16(af0, bf3, acc[0][3]);
    acc[1][3] = WMMA_BF16(af1, bf3, acc[1][3]);
  }
}

__device__ __forceinline__ float row_reduce_sum(float v) {
  __shared__ float red[256];
  int tid = threadIdx.x;
  red[tid] = v;
  __syncthreads();
#pragma unroll
  for (int s = 128; s > 0; s >>= 1) {
    if (tid < s) red[tid] += red[tid + s];
    __syncthreads();
  }
  float r = red[0];
  __syncthreads();
  return r;
}

// ---------------------------------------------------------------------------
// Elementwise / prep kernels
// ---------------------------------------------------------------------------
__global__ void k_f32_to_bf16(const float* __restrict__ src, bf16* __restrict__ dst, int n) {
  int i = blockIdx.x * 256 + threadIdx.x;
  if (i < n) dst[i] = (bf16)src[i];
}

// one block per token: gather embedding + write L2-normalized bf16 copy for sims
__global__ void k_embed(const int* __restrict__ ids, const float* __restrict__ wte,
                        float* __restrict__ x, bf16* __restrict__ xsim) {
  int row = blockIdx.x;            // 0..BT_-1
  int tid = threadIdx.x;
  const float* src = wte + (size_t)ids[row] * E_;
  float* dx = x + (size_t)row * E_;
  float v[3], ss = 0.f;
#pragma unroll
  for (int j = 0; j < 3; ++j) { float f = src[tid + j * 256]; v[j] = f; dx[tid + j * 256] = f; ss += f * f; }
  float tot = row_reduce_sum(ss);
  float inv = 1.0f / (sqrtf(tot) + 1e-8f);
  bf16* ds = xsim + (size_t)row * E_;
#pragma unroll
  for (int j = 0; j < 3; ++j) ds[tid + j * 256] = (bf16)(v[j] * inv);
}

__device__ __forceinline__ void pair_decode(int z, int& b, int& i, int& j) {
  b = z / 28;
  int q = z % 28;
  int ii = 1, t = 0;
  while (q >= t + ii) { t += ii; ++ii; }
  i = ii;
  j = q - t;
}

// sims[z] = cn_i (256x768) @ cn_j^T (768x256)  via WMMA; z enumerates (b,i,j<i)
__global__ void k_gemm_sims(const bf16* __restrict__ xsim, float* __restrict__ simbuf) {
  int z = blockIdx.z;
  int b, i, j;
  pair_decode(z, b, i, j);
  const bf16* A = xsim + (size_t)(b * N_ + i) * C_ * E_;   // current chunk rows
  const bf16* W = xsim + (size_t)(b * N_ + j) * C_ * E_;   // previous chunk rows
  float* D = simbuf + (size_t)z * C_ * C_;
  int wave = threadIdx.x >> 5, lane = threadIdx.x & 31;
  int row0 = blockIdx.x * 64 + (wave & 1) * 32;
  int col0 = (wave >> 1) * 64;
  v8f acc[2][4] = {};
  gemm_core_2x4(A, E_, W, E_, row0, col0, E_, acc);
  int n = lane & 15, mb = (lane >> 4) * 8;
#pragma unroll
  for (int u = 0; u < 2; ++u)
#pragma unroll
    for (int t = 0; t < 4; ++t) {
      int col = col0 + t * 16 + n;
#pragma unroll
      for (int jj = 0; jj < 8; ++jj) {
        int row = row0 + u * 16 + mb + jj;
        D[(size_t)row * C_ + col] = acc[u][t][jj];
      }
    }
}

// scores[b,i,j] = sum_c max_k sims[c,k]
__global__ void k_score(const float* __restrict__ simbuf, float* __restrict__ scores) {
  int z = blockIdx.x;
  int b, i, j;
  pair_decode(z, b, i, j);
  int c = threadIdx.x;
  const float* row = simbuf + ((size_t)z * C_ + c) * C_;
  float best = -1e30f;
  for (int k = 0; k < C_; ++k) best = fmaxf(best, row[k]);
  float tot = row_reduce_sum(best);
  if (c == 0) scores[(b * N_ + i) * N_ + j] = tot;
}

// one thread per (b,i): top-k by score, re-sorted by index, relative weights
__global__ void k_topk(const float* __restrict__ scores, int* __restrict__ sel_idx,
                       float* __restrict__ sel_w) {
  int t = threadIdx.x;
  if (t >= NIMG_) return;
  int b = t / N_, i = t % N_;
  int n_sel = i < 3 ? i : 3;
  float sc[8];
  for (int j = 0; j < i; ++j) sc[j] = scores[(b * N_ + i) * N_ + j];
  bool used[8] = {false, false, false, false, false, false, false, false};
  int   idxs[3] = {-1, -1, -1};
  float vals[3] = {0.f, 0.f, 0.f};
  for (int s = 0; s < n_sel; ++s) {
    float bv = -1e30f; int bj = -1;
    for (int j = 0; j < i; ++j) if (!used[j] && sc[j] > bv) { bv = sc[j]; bj = j; }
    used[bj] = true; idxs[s] = bj; vals[s] = bv;
  }
  for (int a1 = 0; a1 < n_sel; ++a1)
    for (int a2 = a1 + 1; a2 < n_sel; ++a2)
      if (idxs[a2] < idxs[a1]) {
        int ti = idxs[a1]; idxs[a1] = idxs[a2]; idxs[a2] = ti;
        float tv = vals[a1]; vals[a1] = vals[a2]; vals[a2] = tv;
      }
  float denom = (n_sel > 0) ? (vals[0] + 1e-8f) : 1.0f;
  for (int s = 0; s < 3; ++s) {
    sel_idx[(b * N_ + i) * 3 + s] = (s < n_sel) ? idxs[s] : -1;
    sel_w  [(b * N_ + i) * 3 + s] = (s < n_sel) ? vals[s] / denom : 0.f;
  }
}

// grid (1024, 16): assemble extended sequence h (NIMG_, TL_, E_)
__global__ void k_build_h(const float* __restrict__ x, const int* __restrict__ sel_idx,
                          const float* __restrict__ sel_w, float* __restrict__ h) {
  int r   = blockIdx.x;
  int img = blockIdx.y;
  int b = img / N_, i = img % N_;
  int n_sel = i < 3 ? i : 3;
  int pad = TL_ - (n_sel + 1) * C_;
  float* dst = h + ((size_t)img * TL_ + r) * E_;
  int tid = threadIdx.x;
  if (r < pad) {
#pragma unroll
    for (int j = 0; j < 3; ++j) dst[tid + j * 256] = 0.f;
    return;
  }
  int rr = r - pad;
  const float* src;
  float w;
  if (rr < n_sel * C_) {
    int s = rr >> 8, k = rr & 255;
    int jj = sel_idx[img * 3 + s];
    w = sel_w[img * 3 + s];
    src = x + ((size_t)b * 2048 + jj * C_ + k) * E_;
  } else {
    w = 1.0f;
    src = x + ((size_t)b * 2048 + i * C_ + (rr - n_sel * C_)) * E_;
  }
#pragma unroll
  for (int j = 0; j < 3; ++j) dst[tid + j * 256] = src[tid + j * 256] * w;
}

// h (f32) -> xn (bf16) rmsnorm with per-layer weight
__global__ void k_rmsnorm(const float* __restrict__ h, const float* __restrict__ w,
                          bf16* __restrict__ xn) {
  int row = blockIdx.x, tid = threadIdx.x;
  const float* src = h + (size_t)row * E_;
  float v[3], ss = 0.f;
#pragma unroll
  for (int j = 0; j < 3; ++j) { float f = src[tid + j * 256]; v[j] = f; ss += f * f; }
  float tot = row_reduce_sum(ss);
  float inv = rsqrtf(tot / (float)E_ + 1e-10f);
  bf16* d = xn + (size_t)row * E_;
#pragma unroll
  for (int j = 0; j < 3; ++j) d[tid + j * 256] = (bf16)(v[j] * inv * w[tid + j * 256]);
}

// a (f32) -> an (bf16, row-major) and a_t (bf16, transposed per image)
__global__ void k_prep_attn(const float* __restrict__ a, bf16* __restrict__ an,
                            bf16* __restrict__ a_t) {
  int row = blockIdx.x, tid = threadIdx.x;
  const float* src = a + (size_t)row * E_;
  float v[3], ss = 0.f;
#pragma unroll
  for (int j = 0; j < 3; ++j) { float f = src[tid + j * 256]; v[j] = f; ss += f * f; }
  float tot = row_reduce_sum(ss);
  float inv = rsqrtf(tot + 1e-8f);
  int img = row >> 10, r = row & (TL_ - 1);
  bf16* d = an + (size_t)row * E_;
#pragma unroll
  for (int j = 0; j < 3; ++j) {
    int col = tid + j * 256;
    bf16 val = (bf16)(v[j] * inv);
    d[col] = val;
    a_t[((size_t)img * E_ + col) * TL_ + r] = val;
  }
}

// final rows (last C_ of each chunk) -> rmsnorm(lnf) -> fnorm bf16 (BT_, E_)
__global__ void k_final_norm(const float* __restrict__ h, const float* __restrict__ lnf,
                             bf16* __restrict__ fnorm) {
  int row = blockIdx.x;            // b*2048 + t
  int tid = threadIdx.x;
  int b = row >> 11, t = row & 2047;
  int i = t >> 8, k = t & 255;
  const float* src = h + (((size_t)(b * N_ + i) * TL_) + (TL_ - C_) + k) * E_;
  float v[3], ss = 0.f;
#pragma unroll
  for (int j = 0; j < 3; ++j) { float f = src[tid + j * 256]; v[j] = f; ss += f * f; }
  float tot = row_reduce_sum(ss);
  float inv = rsqrtf(tot / (float)E_ + 1e-10f);
  bf16* d = fnorm + (size_t)row * E_;
#pragma unroll
  for (int j = 0; j < 3; ++j) d[tid + j * 256] = (bf16)(v[j] * inv * lnf[tid + j * 256]);
}

// ---------------------------------------------------------------------------
// WMMA GEMM kernels (block = 8 waves as 2Mx4N, block tile 64M x 256N)
// ---------------------------------------------------------------------------
// xp = relu(xn @ ew^T + eb)^2, routed: cols [0,768)->a f32, [768,1536)->b bf16,
// [1536,3072) -> fin[:, 768:2304] bf16
__global__ void k_gemm_enrich(const bf16* __restrict__ xn, const bf16* __restrict__ ew,
                              const float* __restrict__ eb, float* __restrict__ a_buf,
                              bf16* __restrict__ b_buf, bf16* __restrict__ fin) {
  int wave = threadIdx.x >> 5, lane = threadIdx.x & 31;
  int row0 = blockIdx.x * 64 + (wave & 1) * 32;
  int col0 = blockIdx.y * 256 + (wave >> 1) * 64;
  v8f acc[2][4] = {};
  gemm_core_2x4(xn, E_, ew, E_, row0, col0, E_, acc);
  int n = lane & 15, mb = (lane >> 4) * 8;
#pragma unroll
  for (int u = 0; u < 2; ++u)
#pragma unroll
    for (int t = 0; t < 4; ++t) {
      int col = col0 + t * 16 + n;
      float bias = eb[col];
#pragma unroll
      for (int j = 0; j < 8; ++j) {
        int row = row0 + u * 16 + mb + j;
        float y = acc[u][t][j] + bias;
        y = fmaxf(y, 0.f);
        y = y * y;
        if (col < 768)       a_buf[(size_t)row * E_ + col] = y;
        else if (col < 1536) b_buf[(size_t)row * E_ + (col - 768)] = (bf16)y;
        else                 fin[(size_t)row * PROJ_ + (col - 1536 + 768)] = (bf16)y;
      }
    }
}

// cs = an @ an^T per image, masked by tril(spatial_l), stored bf16.
// Blocks entirely above the diagonal skip the K loop and store zeros.
__global__ void k_gemm_cs(const bf16* __restrict__ an, const float* __restrict__ sp,
                          bf16* __restrict__ mcs) {
  int img = blockIdx.z;
  const bf16* A = an + (size_t)img * TL_ * E_;
  bf16* D = mcs + (size_t)img * TL_ * TL_;
  int wave = threadIdx.x >> 5, lane = threadIdx.x & 31;
  int row0 = blockIdx.x * 64 + (wave & 1) * 32;
  int col0 = blockIdx.y * 256 + (wave >> 1) * 64;
  int n = lane & 15, mb = (lane >> 4) * 8;
  if (blockIdx.y * 256 > blockIdx.x * 64 + 63) {   // strictly above diagonal
#pragma unroll
    for (int u = 0; u < 2; ++u)
#pragma unroll
      for (int t = 0; t < 4; ++t) {
        int col = col0 + t * 16 + n;
#pragma unroll
        for (int j = 0; j < 8; ++j)
          D[(size_t)(row0 + u * 16 + mb + j) * TL_ + col] = (bf16)0.f;
      }
    return;
  }
  v8f acc[2][4] = {};
  gemm_core_2x4(A, E_, A, E_, row0, col0, E_, acc);
#pragma unroll
  for (int u = 0; u < 2; ++u)
#pragma unroll
    for (int t = 0; t < 4; ++t) {
      int col = col0 + t * 16 + n;
#pragma unroll
      for (int j = 0; j < 8; ++j) {
        int row = row0 + u * 16 + mb + j;
        float m = (col <= row) ? sp[(size_t)row * TL_ + col] : 0.f;
        D[(size_t)row * TL_ + col] = (bf16)(acc[u][t][j] * m);
      }
    }
}

// a2 = mcs @ a (B operand via a_t). mcs is lower-triangular, so the K loop is
// clamped to the block's max row + 1. Epilogue: fin[:, :768] = bf16(a2 * b).
__global__ void k_gemm_attn(const bf16* __restrict__ mcs, const bf16* __restrict__ a_t,
                            const bf16* __restrict__ b_buf, bf16* __restrict__ fin) {
  int img = blockIdx.z;
  const bf16* A = mcs + (size_t)img * TL_ * TL_;
  const bf16* W = a_t + (size_t)img * E_ * TL_;
  int wave = threadIdx.x >> 5, lane = threadIdx.x & 31;
  int row0 = blockIdx.x * 64 + (wave & 1) * 32;
  int col0 = blockIdx.y * 256 + (wave >> 1) * 64;
  int K_eff = blockIdx.x * 64 + 64;                // cols beyond max row are zero
  if (K_eff > TL_) K_eff = TL_;
  v8f acc[2][4] = {};
  gemm_core_2x4(A, TL_, W, TL_, row0, col0, K_eff, acc);
  int n = lane & 15, mb = (lane >> 4) * 8;
#pragma unroll
  for (int u = 0; u < 2; ++u)
#pragma unroll
    for (int t = 0; t < 4; ++t) {
      int col = col0 + t * 16 + n;
#pragma unroll
      for (int j = 0; j < 8; ++j) {
        int row = row0 + u * 16 + mb + j;
        size_t grow = (size_t)img * TL_ + row;
        float bv = (float)b_buf[grow * E_ + col];
        fin[grow * PROJ_ + col] = (bf16)(acc[u][t][j] * bv);
      }
    }
}

// h += fin @ fw^T + fb
__global__ void k_gemm_fuse(const bf16* __restrict__ fin, const bf16* __restrict__ fw,
                            const float* __restrict__ fb, float* __restrict__ h) {
  int wave = threadIdx.x >> 5, lane = threadIdx.x & 31;
  int row0 = blockIdx.x * 64 + (wave & 1) * 32;
  int col0 = blockIdx.y * 256 + (wave >> 1) * 64;
  v8f acc[2][4] = {};
  gemm_core_2x4(fin, PROJ_, fw, PROJ_, row0, col0, PROJ_, acc);
  int n = lane & 15, mb = (lane >> 4) * 8;
#pragma unroll
  for (int u = 0; u < 2; ++u)
#pragma unroll
    for (int t = 0; t < 4; ++t) {
      int col = col0 + t * 16 + n;
      float bias = fb[col];
#pragma unroll
      for (int j = 0; j < 8; ++j) {
        int row = row0 + u * 16 + mb + j;
        h[(size_t)row * E_ + col] += acc[u][t][j] + bias;
      }
    }
}

// logits = fnorm @ wte^T  -> d_out f32 (BT_, V_)
__global__ void k_gemm_logits(const bf16* __restrict__ fnorm, const bf16* __restrict__ wteb,
                              float* __restrict__ out) {
  int wave = threadIdx.x >> 5, lane = threadIdx.x & 31;
  int row0 = blockIdx.x * 64 + (wave & 1) * 32;
  int col0 = blockIdx.y * 256 + (wave >> 1) * 64;
  v8f acc[2][4] = {};
  gemm_core_2x4(fnorm, E_, wteb, E_, row0, col0, E_, acc);
  int n = lane & 15, mb = (lane >> 4) * 8;
#pragma unroll
  for (int u = 0; u < 2; ++u)
#pragma unroll
    for (int t = 0; t < 4; ++t) {
      int col = col0 + t * 16 + n;
#pragma unroll
      for (int j = 0; j < 8; ++j) {
        int row = row0 + u * 16 + mb + j;
        out[(size_t)row * V_ + col] = acc[u][t][j];
      }
    }
}

// ---------------------------------------------------------------------------
extern "C" void kernel_launch(void* const* d_in, const int* in_sizes, int n_in,
                              void* d_out, int out_size, void* d_ws, size_t ws_size,
                              hipStream_t stream) {
  (void)in_sizes; (void)n_in; (void)out_size; (void)ws_size;
  const int*   ids   = (const int*)d_in[0];
  const float* wte   = (const float*)d_in[1];
  const float* rms_w = (const float*)d_in[2];
  const float* enr_w = (const float*)d_in[3];
  const float* enr_b = (const float*)d_in[4];
  const float* spat  = (const float*)d_in[5];
  const float* fus_w = (const float*)d_in[6];
  const float* fus_b = (const float*)d_in[7];
  const float* lnf_w = (const float*)d_in[8];
  float* out = (float*)d_out;

  char* base = (char*)d_ws;
  size_t off = 0;
  auto take = [&](size_t bytes) -> char* {
    char* p = base + off;
    off += bytes;
    off = (off + 255) & ~(size_t)255;
    return p;
  };

  bf16*  wte_b  = (bf16*)take((size_t)V_ * E_ * 2);
  bf16*  enr_wb = (bf16*)take((size_t)L_ * ED_ * E_ * 2);
  bf16*  fus_wb = (bf16*)take((size_t)L_ * E_ * PROJ_ * 2);
  float* x      = (float*)take((size_t)BT_ * E_ * 4);
  bf16*  xsim   = (bf16*)take((size_t)BT_ * E_ * 2);
  float* simbuf = (float*)take((size_t)NPAIR_ * C_ * C_ * 4);
  float* scores = (float*)take((size_t)NIMG_ * N_ * 4);
  int*   seli   = (int*)take((size_t)NIMG_ * 3 * 4);
  float* selw   = (float*)take((size_t)NIMG_ * 3 * 4);
  float* h      = (float*)take((size_t)ROWS_ * E_ * 4);
  bf16*  xn     = (bf16*)take((size_t)ROWS_ * E_ * 2);
  float* a_buf  = (float*)take((size_t)ROWS_ * E_ * 4);
  bf16*  b_buf  = (bf16*)take((size_t)ROWS_ * E_ * 2);
  bf16*  fin    = (bf16*)take((size_t)ROWS_ * PROJ_ * 2);
  bf16*  an     = (bf16*)take((size_t)ROWS_ * E_ * 2);
  bf16*  a_t    = (bf16*)take((size_t)ROWS_ * E_ * 2);
  bf16*  mcs    = (bf16*)take((size_t)NIMG_ * TL_ * TL_ * 2);
  bf16*  fnorm  = (bf16*)take((size_t)BT_ * E_ * 2);

  // weight conversions to bf16
  {
    int n0 = V_ * E_;
    k_f32_to_bf16<<<(n0 + 255) / 256, 256, 0, stream>>>(wte, wte_b, n0);
    int n1 = L_ * ED_ * E_;
    k_f32_to_bf16<<<(n1 + 255) / 256, 256, 0, stream>>>(enr_w, enr_wb, n1);
    int n2 = L_ * E_ * PROJ_;
    k_f32_to_bf16<<<(n2 + 255) / 256, 256, 0, stream>>>(fus_w, fus_wb, n2);
  }

  // embeddings + normalized copy; WMMA cosine-sim matrix; scores; top-k; assembly
  k_embed<<<BT_, 256, 0, stream>>>(ids, wte, x, xsim);
  k_gemm_sims<<<dim3(C_ / 64, 1, NPAIR_), 256, 0, stream>>>(xsim, simbuf);
  k_score<<<NPAIR_, 256, 0, stream>>>(simbuf, scores);
  k_topk<<<1, 32, 0, stream>>>(scores, seli, selw);
  k_build_h<<<dim3(TL_, NIMG_), 256, 0, stream>>>(x, seli, selw, h);

  // 4 layers
  for (int l = 0; l < L_; ++l) {
    k_rmsnorm<<<ROWS_, 256, 0, stream>>>(h, rms_w + (size_t)l * E_, xn);
    k_gemm_enrich<<<dim3(ROWS_ / 64, ED_ / 256), 256, 0, stream>>>(
        xn, enr_wb + (size_t)l * ED_ * E_, enr_b + (size_t)l * ED_, a_buf, b_buf, fin);
    k_prep_attn<<<ROWS_, 256, 0, stream>>>(a_buf, an, a_t);
    k_gemm_cs<<<dim3(TL_ / 64, TL_ / 256, NIMG_), 256, 0, stream>>>(
        an, spat + (size_t)l * TL_ * TL_, mcs);
    k_gemm_attn<<<dim3(TL_ / 64, E_ / 256, NIMG_), 256, 0, stream>>>(mcs, a_t, b_buf, fin);
    k_gemm_fuse<<<dim3(ROWS_ / 64, E_ / 256), 256, 0, stream>>>(
        fin, fus_wb + (size_t)l * E_ * PROJ_, fus_b + (size_t)l * E_, h);
  }

  // final norm + logits
  k_final_norm<<<BT_, 256, 0, stream>>>(h, lnf_w, fnorm);
  k_gemm_logits<<<dim3(BT_ / 64, V_ / 256), 256, 0, stream>>>(fnorm, wte_b, out);
}
